// WindowAttention_90683939487945
// MI455X (gfx1250) — matrix operand order repaired and checked by reading, not compiled, not run on hardware
//
#include <hip/hip_runtime.h>
#include <math.h>

typedef __bf16 bf16;
typedef bf16  bf16x16 __attribute__((ext_vector_type(16)));
typedef bf16  bf16x8  __attribute__((ext_vector_type(8)));
typedef float v8f     __attribute__((ext_vector_type(8)));
typedef unsigned int u32x4 __attribute__((ext_vector_type(4)));

#define NTOK 49
#define NPAD 64
#define DIMC 256
#define NHEAD 8
#define HDIM 32
#define ATTN_SCALE 0.17677669529663687f  /* 32^-0.5 */

union Frag32B { u32x4 u[2]; bf16x16 v; };

__device__ __forceinline__ v8f zero8() {
  v8f z = {0.f, 0.f, 0.f, 0.f, 0.f, 0.f, 0.f, 0.f};
  return z;
}

__device__ __forceinline__ v8f wmma_bf16(bf16x16 a, bf16x16 b, v8f c) {
  // D = A*B + C, 16x16x32 bf16 -> f32, wave32
  return __builtin_amdgcn_wmma_f32_16x16x32_bf16(
      /*neg_a=*/false, a, /*neg_b=*/false, b,
      /*c_mod=*/(short)0, c, /*reuse_a=*/false, /*reuse_b=*/false);
}

// A-matrix fragment from row-major bf16 storage.
// lane l (0..31): row = given; K elems = kh*16 + e (contiguous 16) with kh=l>>4.
__device__ __forceinline__ bf16x16 load_a_frag(const bf16* base, int stride,
                                               int row, int kbase, int lane) {
  const int kh = (lane >> 4) & 1;
  const bf16* p = base + row * stride + kbase + kh * 16;
  Frag32B f;
  f.u[0] = *reinterpret_cast<const u32x4*>(p);       // K = kbase+kh*16 .. +7
  f.u[1] = *reinterpret_cast<const u32x4*>(p + 8);   // K = kbase+kh*16+8 .. +15
  return f.v;
}

// B-matrix fragment where B's column n lives as a contiguous row in storage
// (e.g. B = K^T with K stored row-major, or B = V with V^T stored row-major).
// lane l: col = given row index; K elems = kh*16 + e (contiguous 16 bf16).
__device__ __forceinline__ bf16x16 load_b_frag(const bf16* base, int stride,
                                               int row, int kbase, int lane) {
  const int kh = (lane >> 4) & 1;
  const bf16* p = base + row * stride + kbase + kh * 16;
  Frag32B f;
  const u32x4* q = reinterpret_cast<const u32x4*>(p);
  f.u[0] = q[0];
  f.u[1] = q[1];
  return f.v;
}

// B fragment from pre-packed weight tiles: tile is 512 bf16, lane takes 32B.
__device__ __forceinline__ bf16x16 load_b_packed(const bf16* tile, int lane) {
  Frag32B f;
  const u32x4* q = reinterpret_cast<const u32x4*>(tile + lane * 16);
  f.u[0] = q[0];
  f.u[1] = q[1];
  return f.v;
}

// Pack W (KxN f32, row-major) into fragment-linear bf16:
// out[((kt*NT+nt)*32 + lane)*16 + e] = W[kt*32 + (lane>>4)*16 + e][nt*16 + (lane&15)]
__global__ void prep_w_kernel(const float* __restrict__ W, int Ncols, int total,
                              bf16* __restrict__ out) {
  int tid = blockIdx.x * 256 + threadIdx.x;
  if (tid >= total) return;
  int e    = tid & 15;
  int lane = (tid >> 4) & 31;
  int t    = tid >> 9;
  int NT   = Ncols >> 4;
  int kt   = t / NT;
  int nt   = t - kt * NT;
  int k = kt * 32 + (lane >> 4) * 16 + e;
  int n = nt * 16 + (lane & 15);
  out[tid] = (bf16)W[k * Ncols + n];
}

__global__ __launch_bounds__(256, 4) void winattn_kernel(
    const float* __restrict__ xG, const float* __restrict__ maskG,
    const float* __restrict__ qkvB, const float* __restrict__ projB,
    const float* __restrict__ tableG,
    const bf16* __restrict__ qkvW, const bf16* __restrict__ projW,
    float* __restrict__ outG) {
  // Dynamic LDS: 81920 bf16 = 160KB -> 2 workgroups / WGP
  extern __shared__ __align__(16) bf16 smem[];
  bf16* sX  = smem;          // 64x256   (phase 1 only, overlaps sP)
  bf16* sP  = smem;          // 8x64x64  (phase 2+)
  bf16* sQK = smem + 32768;  // 64x512 rows: q cols 0..255, k cols 256..511
  bf16* sO  = smem + 32768;  // 64x256   (phase 3+, overlaps sQK)
  bf16* sVT = smem + 65536;  // 256x64   V transposed: [dim][token]

  const int tid  = threadIdx.x;
  const int lane = tid & 31;
  // Force wave id into an SGPR: makes per-wave tile indices / branches scalar.
  const int wave = __builtin_amdgcn_readfirstlane(tid >> 5);  // 0..7
  const int kh   = lane >> 4;
  const int cn   = lane & 15;
  const int win  = blockIdx.x;
  const float NEG_INF = -__builtin_inff();

  // ---- Phase 0: stage x -> LDS bf16, 8 elts/thread/iter (zero-pad rows 49..63) ----
  const float* xp = xG + (size_t)win * (NTOK * DIMC);
  for (int c = tid; c < (NPAD * DIMC) / 8; c += 256) {
    int row  = c >> 5;           // 32 chunks of 8 per 256-wide row
    int col8 = (c & 31) * 8;
    bf16x8 v;
    if (row < NTOK) {
      const float* s = xp + row * DIMC + col8;
#pragma unroll
      for (int q = 0; q < 8; ++q) v[q] = (bf16)s[q];
    } else {
#pragma unroll
      for (int q = 0; q < 8; ++q) v[q] = (bf16)0.0f;
    }
    *reinterpret_cast<bf16x8*>(sX + row * DIMC + col8) = v;
  }
  __syncthreads();

  // ---- Phase 1: qkv = x @ qkv_w + qkv_b  (64x768, 48 col-tiles / 8 waves) ----
  // j outer: the 8 B weight-fragments are mt-invariant -> load once per j.
  for (int j = 0; j < 6; ++j) {
    const int nt  = wave * 6 + j;            // scalar
    const int col = nt * 16 + cn;
    bf16x16 bfr[8];
#pragma unroll
    for (int kt = 0; kt < 8; ++kt)
      bfr[kt] = load_b_packed(qkvW + ((kt * 48 + nt) << 9), lane);
    const float bias = qkvB[col];
    for (int mt = 0; mt < 4; ++mt) {
      v8f acc = zero8();
#pragma unroll
      for (int kt = 0; kt < 8; ++kt) {
        bf16x16 a = load_a_frag(sX, DIMC, mt * 16 + cn, kt * 32, lane);
        acc = wmma_bf16(a, bfr[kt], acc);
      }
      // nt is scalar -> s_cmp/s_cbranch, no exec juggling
      if (nt < 32) {  // q | k, row-major
        bf16* dst = sQK + (mt * 16 + kh * 8) * 512 + col;
#pragma unroll
        for (int r = 0; r < 8; ++r) dst[r * 512] = (bf16)(acc[r] + bias);
      } else {        // v, transposed [dim][token]: 8 contiguous -> ds_store_b128
        bf16* dst = sVT + (col - 512) * NPAD + mt * 16 + kh * 8;
#pragma unroll
        for (int r = 0; r < 8; ++r) dst[r] = (bf16)(acc[r] + bias);
      }
    }
  }
  __syncthreads();

  // ---- Phase 2: per-head scores + bias + mask + double softmax -> sP ----
  {
    const int h = wave;                       // scalar head id
    const float* maskp = maskG + (size_t)(win & 63) * (NTOK * NTOK);
    bf16* sPh = sP + h * (NPAD * NPAD);
    for (int mt = 0; mt < 4; ++mt) {
      v8f acc[4];
      bf16x16 aq = load_a_frag(sQK, 512, mt * 16 + cn, h * HDIM, lane);
#pragma unroll
      for (int nt = 0; nt < 4; ++nt) {
        bf16x16 bk = load_b_frag(sQK, 512, nt * 16 + cn, 256 + h * HDIM, lane);
        acc[nt] = wmma_bf16(aq, bk, zero8());
      }
      // scale + relative-position bias + mask; -inf outside 49x49
#pragma unroll
      for (int nt = 0; nt < 4; ++nt) {
        int jc = nt * 16 + cn;
#pragma unroll
        for (int r = 0; r < 8; ++r) {
          int ic = mt * 16 + r + kh * 8;
          float v = acc[nt][r] * ATTN_SCALE;
          if (ic < NTOK && jc < NTOK) {
            int d0 = (ic % 7) - (jc % 7) + 6;
            int d1 = (ic / 7) - (jc / 7) + 6;
            v += tableG[(d0 * 13 + d1) * NHEAD + h] + maskp[ic * NTOK + jc];
          } else {
            v = NEG_INF;
          }
          acc[nt][r] = v;
        }
      }
      // softmax over j, applied twice (faithful to reference)
      for (int pass = 0; pass < 2; ++pass) {
#pragma unroll
        for (int r = 0; r < 8; ++r) {
          float m = fmaxf(fmaxf(acc[0][r], acc[1][r]),
                          fmaxf(acc[2][r], acc[3][r]));
          for (int s = 1; s < 16; s <<= 1) m = fmaxf(m, __shfl_xor(m, s, 32));
          float sum = 0.f;
#pragma unroll
          for (int nt = 0; nt < 4; ++nt) {
            float e = __expf(acc[nt][r] - m);
            acc[nt][r] = e;
            sum += e;
          }
          for (int s = 1; s < 16; s <<= 1) sum += __shfl_xor(sum, s, 32);
          float inv = 1.0f / sum;
#pragma unroll
          for (int nt = 0; nt < 4; ++nt) acc[nt][r] *= inv;
        }
        if (pass == 0) {  // re-mask padded columns for the 2nd softmax
#pragma unroll
          for (int nt = 0; nt < 4; ++nt)
            if (nt * 16 + cn >= NTOK)
#pragma unroll
              for (int r = 0; r < 8; ++r) acc[nt][r] = NEG_INF;
        }
      }
      // store P (zero padded rows so P@V stays NaN-free)
#pragma unroll
      for (int nt = 0; nt < 4; ++nt) {
        bf16* dst = sPh + (mt * 16 + kh * 8) * NPAD + nt * 16 + cn;
#pragma unroll
        for (int r = 0; r < 8; ++r) {
          int ic = mt * 16 + r + kh * 8;
          dst[r * NPAD] = (ic < NTOK) ? (bf16)acc[nt][r] : (bf16)0.0f;
        }
      }
    }
  }
  __syncthreads();

  // ---- Phase 3: O = P @ V per head -> sO (bf16, row-major 64x256) ----
  {
    const int h = wave;
    const bf16* sPh = sP + h * (NPAD * NPAD);
    for (int mt = 0; mt < 4; ++mt) {
#pragma unroll
      for (int ntv = 0; ntv < 2; ++ntv) {
        v8f acc = zero8();
#pragma unroll
        for (int kt = 0; kt < 2; ++kt) {
          bf16x16 a = load_a_frag(sPh, NPAD, mt * 16 + cn, kt * 32, lane);
          bf16x16 b = load_b_frag(sVT, NPAD, h * HDIM + ntv * 16 + cn,
                                  kt * 32, lane);
          acc = wmma_bf16(a, b, acc);
        }
        bf16* dst = sO + (mt * 16 + kh * 8) * DIMC + h * HDIM + ntv * 16 + cn;
#pragma unroll
        for (int r = 0; r < 8; ++r) dst[r * DIMC] = (bf16)acc[r];
      }
    }
  }
  __syncthreads();

  // ---- Phase 4: out = O @ proj_w + proj_b, store rows < 49 (f32) ----
  for (int t = wave; t < 64; t += 8) {
    int mt = t & 3;    // scalar
    int nt = t >> 2;   // scalar
    v8f acc = zero8();
#pragma unroll
    for (int kt = 0; kt < 8; ++kt) {
      bf16x16 a = load_a_frag(sO, DIMC, mt * 16 + cn, kt * 32, lane);
      bf16x16 b = load_b_packed(projW + ((kt * 16 + nt) << 9), lane);
      acc = wmma_bf16(a, b, acc);
    }
    int col = nt * 16 + cn;
    float pb = projB[col];
    float* dst = outG + ((size_t)win * NTOK + mt * 16 + kh * 8) * DIMC + col;
#pragma unroll
    for (int r = 0; r < 8; ++r) {
      int row = mt * 16 + r + kh * 8;
      if (row < NTOK) dst[r * DIMC] = acc[r] + pb;
    }
  }
}

extern "C" void kernel_launch(void* const* d_in, const int* in_sizes, int n_in,
                              void* d_out, int out_size, void* d_ws, size_t ws_size,
                              hipStream_t stream) {
  const float* x      = (const float*)d_in[0];
  const float* mask   = (const float*)d_in[1];
  const float* qkv_w  = (const float*)d_in[2];  // 256x768
  const float* qkv_b  = (const float*)d_in[3];
  const float* proj_w = (const float*)d_in[4];  // 256x256
  const float* proj_b = (const float*)d_in[5];
  const float* table  = (const float*)d_in[6];  // 169x8
  float* out = (float*)d_out;

  // Workspace: fragment-packed bf16 weights (shared by all 4096 blocks, L2-resident)
  bf16* qkvW  = (bf16*)d_ws;           // 256*768 elts = 384KB
  bf16* projW = qkvW + 256 * 768;      // 256*256 elts = 128KB

  prep_w_kernel<<<(256 * 768 + 255) / 256, 256, 0, stream>>>(qkv_w, 768,
                                                             256 * 768, qkvW);
  prep_w_kernel<<<(256 * 256 + 255) / 256, 256, 0, stream>>>(proj_w, 256,
                                                             256 * 256, projW);

  const size_t shmem = 81920 * sizeof(bf16);  // 160KB dynamic LDS
  hipFuncSetAttribute((const void*)winattn_kernel,
                      hipFuncAttributeMaxDynamicSharedMemorySize, (int)shmem);
  winattn_kernel<<<4096, 256, shmem, stream>>>(x, mask, qkv_b, proj_b, table,
                                               qkvW, projW, out);
}
